// CustomModel_6244882448451
// MI455X (gfx1250) — compile-verified, matrix operand order checked by reading, not run on hardware
//
#include <hip/hip_runtime.h>
#include <hip/hip_bf16.h>
#include <math.h>

// ---------------------------------------------------------------------------
// GAT (2-layer, H=1) for MI455X / gfx1250.
//   N=100000 nodes, E=1600000 edges, F=64, C=64, O=32, ED=16, M=20000.
// Dense node GEMMs: v_wmma_f32_16x16x32_bf16, fed by pre-converted bf16 X
// (two global_load_b128 per A fragment) and pre-swizzled bf16 W fragments
// (two global_load_b128 per B fragment) -- no cvts in the inner loop.
// Edge-attention term collapses to edge_attr @ (We @ a_e) (length-16 dot).
// Segment softmax: uint-key atomicMax, fp32 atomicAdd; aggregation is
// wave32-per-edge scatter with global_atomic_add_f32.
// Workspace requirement: ~86 MB (layout in kernel_launch).
// ---------------------------------------------------------------------------

typedef __attribute__((ext_vector_type(16))) __bf16 v16bf;
typedef __attribute__((ext_vector_type(8)))  __bf16 v8bf;
typedef __attribute__((ext_vector_type(8)))  float  v8f;

#define GAT_N   100000
#define GAT_E   1600000
#define GAT_ED  16
#define GAT_M   20000

// ---------------- float <-> monotonic uint key (for atomic max) ------------
__device__ __forceinline__ unsigned f2key(float f) {
  unsigned u = __float_as_uint(f);
  return (u & 0x80000000u) ? ~u : (u | 0x80000000u);
}
__device__ __forceinline__ float key2f(unsigned k) {
  unsigned u = (k & 0x80000000u) ? (k & 0x7fffffffu) : ~k;
  return __uint_as_float(u);
}

// ---------------- f32 -> bf16 row-major convert ----------------------------
__global__ void cvt_bf16_kernel(const float* __restrict__ in,
                                __bf16* __restrict__ out, int total) {
  int i = blockIdx.x * blockDim.x + threadIdx.x;
  if (i < total) out[i] = (__bf16)in[i];
}

// ---------------- repack W[64,NCOLS] into per-lane WMMA B fragments --------
// wfrag[((kb*NT + t)*32 + lane)*16 + j] = bf16( W[k(lane,j,kb)][t*16 + lane%16] )
__global__ void repack_w_kernel(const float* __restrict__ W,
                                __bf16* __restrict__ wfrag, int ncols) {
  int nt = ncols / 16;
  int total = 2 * nt * 32 * 16;
  int i = blockIdx.x * blockDim.x + threadIdx.x;
  if (i >= total) return;
  int j    = i & 15;
  int lane = (i >> 4) & 31;
  int t    = (i >> 9) % nt;
  int kb   = i / (512 * nt);
  int half = lane >> 4, lcol = lane & 15;
  int p    = j >> 1;
  int koff = ((p < 4) ? 2 * p : 16 + 2 * (p - 4)) + (j & 1);
  int k    = kb * 32 + half * 8 + koff;
  wfrag[i] = (__bf16)W[(size_t)k * ncols + t * 16 + lcol];
}

// ---------------- Y[n_rows, NCOLS] = X[n_rows, 64] @ W[64, NCOLS] ----------
// One wave per 16-row block; NCOLS/16 accumulators; bf16 WMMA, fp32 acc.
template <int NCOLS>
__global__ void gat_gemm_wmma(const __bf16* __restrict__ X,
                              const __bf16* __restrict__ wfrag,
                              float* __restrict__ Y, int n_rows) {
  constexpr int K  = 64;
  constexpr int NT = NCOLS / 16;
  int wave = (blockIdx.x * blockDim.x + threadIdx.x) >> 5;
  int lane = threadIdx.x & 31;
  if (wave >= (n_rows >> 4)) return;   // n_rows multiple of 16
  int half = lane >> 4;
  int lrow = lane & 15;

  v8f acc[NT] = {};
  const __bf16* xr = X + (size_t)(wave * 16 + lrow) * K;

#pragma unroll
  for (int kb = 0; kb < K / 32; ++kb) {
    // A fragment: two contiguous 16B runs (K pairs 0-7 and 16-23, +half*8)
    v8bf alo = *(const v8bf*)(xr + kb * 32 + half * 8);
    v8bf ahi = *(const v8bf*)(xr + kb * 32 + half * 8 + 16);
    v16bf a = __builtin_shufflevector(alo, ahi, 0, 1, 2, 3, 4, 5, 6, 7,
                                      8, 9, 10, 11, 12, 13, 14, 15);
#pragma unroll
    for (int t = 0; t < NT; ++t) {
      const __bf16* wf = wfrag + ((size_t)((kb * NT + t) * 32 + lane)) * 16;
      v8bf blo = *(const v8bf*)(wf);
      v8bf bhi = *(const v8bf*)(wf + 8);
      v16bf b = __builtin_shufflevector(blo, bhi, 0, 1, 2, 3, 4, 5, 6, 7,
                                        8, 9, 10, 11, 12, 13, 14, 15);
      acc[t] = __builtin_amdgcn_wmma_f32_16x16x32_bf16(
          false, a, false, b, (short)0, acc[t], false, false);
    }
  }
  // store C/D: VGPR r -> row r (lanes 0-15) / row 8+r (lanes 16-31)
#pragma unroll
  for (int t = 0; t < NT; ++t) {
#pragma unroll
    for (int r = 0; r < 8; ++r) {
      int orow = wave * 16 + r + half * 8;
      Y[(size_t)orow * NCOLS + t * 16 + lrow] = acc[t][r];
    }
  }
}

// ---------------- wv[d] = sum_c We[d,c] * a_e[c] ---------------------------
__global__ void edge_vec_kernel(const float* __restrict__ We,
                                const float* __restrict__ a_e,
                                float* __restrict__ wv, int cout) {
  int d = threadIdx.x;
  if (d < GAT_ED) {
    float s = 0.f;
    for (int c = 0; c < cout; ++c) s += We[d * cout + c] * a_e[c];
    wv[d] = s;
  }
}

// ---------------- per-node attention scalars (wave per node) ---------------
template <int COUT>
__global__ void node_alpha_kernel(const float* __restrict__ hx,
                                  const float* __restrict__ a_s,
                                  const float* __restrict__ a_d,
                                  float* __restrict__ alpS,
                                  float* __restrict__ alpD, int n) {
  int wave = (blockIdx.x * blockDim.x + threadIdx.x) >> 5;
  int lane = threadIdx.x & 31;
  if (wave >= n) return;
  const float* r = hx + (size_t)wave * COUT;
  float ss = 0.f, sd = 0.f;
#pragma unroll
  for (int c = lane; c < COUT; c += 32) {
    float v = r[c];
    ss += v * a_s[c];
    sd += v * a_d[c];
  }
#pragma unroll
  for (int off = 16; off > 0; off >>= 1) {
    ss += __shfl_xor(ss, off, 32);
    sd += __shfl_xor(sd, off, 32);
  }
  if (lane == 0) { alpS[wave] = ss; alpD[wave] = sd; }
}

// ---------------- init: agg = bias (broadcast), mx keys, denom -------------
__global__ void init_kernel(float* __restrict__ agg, const float* __restrict__ b,
                            unsigned* __restrict__ mxkey, float* __restrict__ denom,
                            int n, int cout) {
  int i = blockIdx.x * blockDim.x + threadIdx.x;
  if (i < n * cout) agg[i] = b[i & (cout - 1)];
  if (i < n) { mxkey[i] = 0u; denom[i] = 0.f; }
}

// ---------------- edge alpha (leaky relu) + segment max --------------------
__global__ void edge_alpha_kernel(const int* __restrict__ src,
                                  const int* __restrict__ dst,
                                  const float* __restrict__ edge_attr,
                                  const float* __restrict__ wv,
                                  const float* __restrict__ alpS,
                                  const float* __restrict__ alpD,
                                  float* __restrict__ alpha_out,
                                  unsigned* __restrict__ mxkey, int n) {
  int e = blockIdx.x * blockDim.x + threadIdx.x;
  if (e >= GAT_E + n) return;
  int s, d;
  float ea = 0.f;
  if (e < GAT_E) {
    s = src[e];
    d = dst[e];
    const float* ar = edge_attr + (size_t)e * GAT_ED;
#pragma unroll
    for (int j = 0; j < GAT_ED; ++j) ea += ar[j] * wv[j];
  } else {
    s = d = e - GAT_E;              // self loop, zero edge attr
  }
  float raw = alpS[s] + alpD[d] + ea;
  float lr = raw > 0.f ? raw : 0.2f * raw;
  alpha_out[e] = lr;
  atomicMax(mxkey + d, f2key(lr));
}

// ---------------- exp(alpha - max) + segment sum ---------------------------
__global__ void edge_exp_kernel(const int* __restrict__ dst,
                                float* __restrict__ exv,
                                const unsigned* __restrict__ mxkey,
                                float* __restrict__ denom, int n) {
  int e = blockIdx.x * blockDim.x + threadIdx.x;
  if (e >= GAT_E + n) return;
  int d = (e < GAT_E) ? dst[e] : (e - GAT_E);
  float v = expf(exv[e] - key2f(mxkey[d]));
  exv[e] = v;
  atomicAdd(denom + d, v);
}

// ---------------- wave32 per edge: agg[dst] += coef * hx[src] --------------
template <int COUT>
__global__ void edge_agg_kernel(const int* __restrict__ src,
                                const int* __restrict__ dst,
                                const float* __restrict__ exv,
                                const float* __restrict__ denom,
                                const float* __restrict__ hx,
                                float* __restrict__ agg, int n) {
  int wave = (blockIdx.x * blockDim.x + threadIdx.x) >> 5;
  int lane = threadIdx.x & 31;
  if (wave >= GAT_E + n) return;
  int s = (wave < GAT_E) ? src[wave] : (wave - GAT_E);
  int d = (wave < GAT_E) ? dst[wave] : (wave - GAT_E);
  float coef = exv[wave] / denom[d];
  const float* hr = hx + (size_t)s * COUT;
  float* ar = agg + (size_t)d * COUT;
#pragma unroll
  for (int c = lane; c < COUT; c += 32) atomicAdd(ar + c, coef * hr[c]);
}

// ---------------- ELU (f32 out, and bf16 out for layer-1 -> layer-2) -------
__global__ void elu_f32_kernel(const float* __restrict__ in,
                               float* __restrict__ out, int total) {
  int i = blockIdx.x * blockDim.x + threadIdx.x;
  if (i >= total) return;
  float v = in[i];
  out[i] = v > 0.f ? v : (expf(v) - 1.f);
}
__global__ void elu_bf16_kernel(const float* __restrict__ in,
                                __bf16* __restrict__ out, int total) {
  int i = blockIdx.x * blockDim.x + threadIdx.x;
  if (i >= total) return;
  float v = in[i];
  out[i] = (__bf16)(v > 0.f ? v : (expf(v) - 1.f));
}

// ---------------- final mask gather ----------------------------------------
__global__ void gather_kernel(const float* __restrict__ h2,
                              const int* __restrict__ mask,
                              float* __restrict__ out) {
  int i = blockIdx.x * blockDim.x + threadIdx.x;
  if (i >= GAT_M * 32) return;
  int m = i >> 5, c = i & 31;           // cout == 32
  out[i] = h2[(size_t)mask[m] * 32 + c];
}

// ---------------- one GAT layer (up to but excluding the ELU) --------------
template <int COUT>
static void run_gat_layer(const __bf16* xin, const int* src, const int* dst,
                          const float* edge_attr, const float* W,
                          const float* a_s, const float* a_d, const float* We,
                          const float* a_e, const float* b, __bf16* wfrag,
                          float* hx, float* agg, float* alpS, float* alpD,
                          unsigned* mxkey, float* denom, float* exv, float* wv,
                          hipStream_t stream) {
  const int n = GAT_N, tot = GAT_E + GAT_N;
  edge_vec_kernel<<<1, 32, 0, stream>>>(We, a_e, wv, COUT);
  repack_w_kernel<<<(2 * (COUT / 16) * 512 + 255) / 256, 256, 0, stream>>>(W, wfrag, COUT);
  {
    int waves = n / 16;                 // 6250
    gat_gemm_wmma<COUT><<<(waves * 32 + 255) / 256, 256, 0, stream>>>(xin, wfrag, hx, n);
  }
  node_alpha_kernel<COUT><<<(n * 32 + 255) / 256, 256, 0, stream>>>(hx, a_s, a_d, alpS, alpD, n);
  init_kernel<<<(n * COUT + 255) / 256, 256, 0, stream>>>(agg, b, mxkey, denom, n, COUT);
  edge_alpha_kernel<<<(tot + 255) / 256, 256, 0, stream>>>(src, dst, edge_attr, wv,
                                                           alpS, alpD, exv, mxkey, n);
  edge_exp_kernel<<<(tot + 255) / 256, 256, 0, stream>>>(dst, exv, mxkey, denom, n);
  edge_agg_kernel<COUT><<<(tot * 32 + 255) / 256, 256, 0, stream>>>(src, dst, exv,
                                                                    denom, hx, agg, n);
}

// ---------------------------------------------------------------------------
extern "C" void kernel_launch(void* const* d_in, const int* in_sizes, int n_in,
                              void* d_out, int out_size, void* d_ws, size_t ws_size,
                              hipStream_t stream) {
  const float* x         = (const float*)d_in[0];
  const int*   edge_idx  = (const int*)d_in[1];
  const int*   mask      = (const int*)d_in[2];
  const float* edge_attr = (const float*)d_in[3];
  const float* W1   = (const float*)d_in[4];
  const float* a_s1 = (const float*)d_in[5];
  const float* a_d1 = (const float*)d_in[6];
  const float* We1  = (const float*)d_in[7];
  const float* a_e1 = (const float*)d_in[8];
  const float* b1   = (const float*)d_in[9];
  const float* W2   = (const float*)d_in[10];
  const float* a_s2 = (const float*)d_in[11];
  const float* a_d2 = (const float*)d_in[12];
  const float* We2  = (const float*)d_in[13];
  const float* a_e2 = (const float*)d_in[14];
  const float* b2   = (const float*)d_in[15];
  float* out = (float*)d_out;

  const int* src = edge_idx;
  const int* dst = edge_idx + GAT_E;

  // ---- workspace layout (float units); total ~21.4M floats (~86 MB) ----
  float* ws = (float*)d_ws;
  size_t off = 0;
  float*  hx    = ws + off; off += (size_t)GAT_N * 64;       // layer hx (f32, reused)
  float*  agg   = ws + off; off += (size_t)GAT_N * 64;       // aggregation (reused)
  __bf16* xbf   = (__bf16*)(ws + off); off += (size_t)GAT_N * 32;  // bf16 x
  __bf16* h1bf  = (__bf16*)(ws + off); off += (size_t)GAT_N * 32;  // bf16 layer-1 out
  float*  alpS  = ws + off; off += GAT_N;
  float*  alpD  = ws + off; off += GAT_N;
  unsigned* mxkey = (unsigned*)(ws + off); off += GAT_N;
  float*  denom = ws + off; off += GAT_N;
  float*  exv   = ws + off; off += (size_t)(GAT_E + GAT_N);
  float*  wv    = ws + off; off += 64;
  __bf16* wfrag = (__bf16*)(ws + off); off += 2048;          // 4096 bf16 max

  // ---- layer 1: F=64 -> C=64 ----
  cvt_bf16_kernel<<<(GAT_N * 64 + 255) / 256, 256, 0, stream>>>(x, xbf, GAT_N * 64);
  run_gat_layer<64>(xbf, src, dst, edge_attr, W1, a_s1, a_d1, We1, a_e1, b1,
                    wfrag, hx, agg, alpS, alpD, mxkey, denom, exv, wv, stream);
  elu_bf16_kernel<<<(GAT_N * 64 + 255) / 256, 256, 0, stream>>>(agg, h1bf, GAT_N * 64);

  // ---- layer 2: C=64 -> O=32 (reuse hx/agg; f32 result in upper agg) ----
  run_gat_layer<32>(h1bf, src, dst, edge_attr, W2, a_s2, a_d2, We2, a_e2, b2,
                    wfrag, hx, agg, alpS, alpD, mxkey, denom, exv, wv, stream);
  float* hout2 = agg + (size_t)GAT_N * 32;
  elu_f32_kernel<<<(GAT_N * 32 + 255) / 256, 256, 0, stream>>>(agg, hout2, GAT_N * 32);

  // ---- gather masked rows: out[M, 32] ----
  gather_kernel<<<(GAT_M * 32 + 255) / 256, 256, 0, stream>>>(hout2, mask, out);
}
